// RnnLayer_44152263802968
// MI455X (gfx1250) — matrix operand order repaired and checked by reading, not compiled, not run on hardware
//
#include <hip/hip_runtime.h>
#include <hip/hip_bf16.h>
#include <math.h>

// ---------------------------------------------------------------------------
// 5-layer stacked LSTM on MI455X (gfx1250), bf16 WMMA + f32 accumulate.
//   T=512, B=256, IN=64, H=128, L=5   (fp32 in / fp32 out)
//
// All matrix operands live in WMMA-fragment-native layouts in the workspace:
//   * weights:   bf16 B-fragments  (one-time prep kernels)
//   * activations: bf16 A-fragments (produced directly by the recurrence)
//   * pre = x@W_ih^T + b : f32 in C-fragment-swizzled layout, so each
//     (t, batch-tile) chunk is one contiguous 32KB block.
// The recurrence double-buffers that 32KB chunk in LDS via the Tensor Data
// Mover (tensor_load_to_lds, TENSORcnt), prefetching step t+1 during step t.
// Gate nonlinearities use native v_exp_f32 / v_rcp_f32 (latency-critical).
//
// Workspace (assumed ws_size >= ~322 MB):
//   [0,256MB)   pre_sw  f32  [8192 tiles][8 waves][4 nt][32 lanes][8]
//   [+32MB]     afA     u32  [8192][4 kt][32][8]   (bf16 pair A-fragments)
//   [+32MB]     afB     u32  same
//   [+~1.2MB]   wf_ih0, wf_ih1..4, wf_hh0..4 (bf16 B-fragments)
// ---------------------------------------------------------------------------

#define TT 512
#define BB 256
#define HH 128
#define GH 512                 // 4*H
#define TILES (TT * BB / 16)   // 8192

typedef __attribute__((ext_vector_type(16))) __bf16 v16bf;
typedef __attribute__((ext_vector_type(8)))  float  v8f;
typedef __attribute__((ext_vector_type(4)))  unsigned int u32x4;
typedef __attribute__((ext_vector_type(8)))  int i32x8;
typedef __attribute__((ext_vector_type(4)))  int i32x4;

#if defined(__HIP_DEVICE_COMPILE__) && defined(__has_builtin)
#if __has_builtin(__builtin_amdgcn_tensor_load_to_lds) && \
    __has_builtin(__builtin_amdgcn_s_wait_tensorcnt)
#define HAVE_TDM 1
#endif
#if __has_builtin(__builtin_amdgcn_exp2f) && __has_builtin(__builtin_amdgcn_rcpf)
#define HAVE_FASTMATH 1
#endif
#endif
#ifndef HAVE_TDM
#define HAVE_TDM 0
#endif
#ifndef HAVE_FASTMATH
#define HAVE_FASTMATH 0
#endif

union BF16x16 {
  v16bf v;
  unsigned int u32[8];
  uint4 q[2];
};

// f32 -> bf16 RNE, pack two into one dword (lo in [15:0]).
__device__ __forceinline__ unsigned int pack2bf(float lo, float hi) {
  unsigned int a = __float_as_uint(lo);
  unsigned int b = __float_as_uint(hi);
  a += 0x7FFFu + ((a >> 16) & 1u);
  b += 0x7FFFu + ((b >> 16) & 1u);
  return (a >> 16) | (b & 0xFFFF0000u);
}

#define LOG2E 1.4426950408889634f

// sigmoid via native v_exp_f32 + v_rcp_f32: saturates correctly at +-inf.
__device__ __forceinline__ float sigmoidf_(float x) {
#if HAVE_FASTMATH
  return __builtin_amdgcn_rcpf(1.0f + __builtin_amdgcn_exp2f(-x * LOG2E));
#else
  return 1.0f / (1.0f + __expf(-x));
#endif
}

__device__ __forceinline__ float tanhf_(float x) {
#if HAVE_FASTMATH
  // tanh(x) = 2*sigmoid(2x) - 1
  const float s =
      __builtin_amdgcn_rcpf(1.0f + __builtin_amdgcn_exp2f(-2.0f * LOG2E * x));
  return 2.0f * s - 1.0f;
#else
  return tanhf(x);
#endif
}

// A-fragment dword v (lane half h) covers K = (v>=4)*16 + h*8 + (v%4)*2 (+1).
__device__ __forceinline__ int a_koff(int v, int half) {
  return ((v >> 2) << 4) + (half << 3) + ((v & 3) << 1);
}

// ---------------------------------------------------------------------------
// One-time weight prep: W[g,k] f32 -> bf16 B-fragments.
// Layout: wf[((wave*4+nt)*KT + kt)*32 + lane][8].  grid.x = K/32, block = 256.
// ---------------------------------------------------------------------------
__global__ __launch_bounds__(256) void prep_wfrag_kernel(
    const float* __restrict__ W, int K, unsigned int* __restrict__ wf) {
  const int lane = threadIdx.x & 31;
  const int wave = threadIdx.x >> 5;
  const int r = lane & 15, half = lane >> 4;
  const int kt = blockIdx.x;
  const int KT = gridDim.x;
#pragma unroll
  for (int nt = 0; nt < 4; ++nt) {
    const int g = wave * 64 + nt * 16 + r;
    const float* wrow = W + (size_t)g * K;
    unsigned int o[8];
#pragma unroll
    for (int j = 0; j < 8; ++j) {  // B dword j: K = (j>=4)*16 + half*8 + (j%4)*2 (+1)
      const int k = kt * 32 + a_koff(j, half);
      o[j] = pack2bf(wrow[k], wrow[k + 1]);
    }
    uint4* dst = (uint4*)(wf + (((size_t)(wave * 4 + nt) * KT + kt) * 32 + lane) * 8);
    dst[0] = make_uint4(o[0], o[1], o[2], o[3]);
    dst[1] = make_uint4(o[4], o[5], o[6], o[7]);
  }
}

// Layer-0 input: x f32 [T*B,64] -> bf16 A-fragments. grid = 8192, block = 64.
__global__ __launch_bounds__(64) void x_to_afrag_kernel(
    const float* __restrict__ x, unsigned int* __restrict__ af) {
  const int lane = threadIdx.x & 31;
  const int kt = threadIdx.x >> 5;  // 0..1
  const int r = lane & 15, half = lane >> 4;
  const int tile = blockIdx.x;
  const float* xrow = x + ((size_t)tile * 16 + r) * 64;
  unsigned int o[8];
#pragma unroll
  for (int v = 0; v < 8; ++v) {
    const int k = kt * 32 + a_koff(v, half);
    o[v] = pack2bf(xrow[k], xrow[k + 1]);
  }
  uint4* dst = (uint4*)(af + ((size_t)(tile * 2 + kt) * 32 + lane) * 8);
  dst[0] = make_uint4(o[0], o[1], o[2], o[3]);
  dst[1] = make_uint4(o[4], o[5], o[6], o[7]);
}

// ---------------------------------------------------------------------------
// pre_sw[tile][wave][nt][lane][8] = af(tile) @ wf^T + b_ih + b_hh
// grid.x = 8192 tiles, block = 256 (8 waves; wave w owns gate cols [64w,+64)).
// ---------------------------------------------------------------------------
template <int D>
__global__ __launch_bounds__(256) void pre_gemm_kernel(
    const unsigned int* __restrict__ af, const unsigned int* __restrict__ wf,
    const float* __restrict__ b_ih, const float* __restrict__ b_hh,
    float* __restrict__ pre_sw) {
  constexpr int KT = D / 32;
  const int lane = threadIdx.x & 31;
  const int wave = threadIdx.x >> 5;
  const int r = lane & 15;
  const int tile = blockIdx.x;

  __builtin_prefetch(wf, 0, 1);  // global_prefetch_b8

  BF16x16 afrag[KT];
#pragma unroll
  for (int kt = 0; kt < KT; ++kt) {
    const uint4* p = (const uint4*)(af + ((size_t)(tile * KT + kt) * 32 + lane) * 8);
    afrag[kt].q[0] = p[0];
    afrag[kt].q[1] = p[1];
  }

#pragma unroll
  for (int nt = 0; nt < 4; ++nt) {
    const int g = wave * 64 + nt * 16 + r;
    const float bias = b_ih[g] + b_hh[g];
    v8f acc;
#pragma unroll
    for (int v = 0; v < 8; ++v) acc[v] = bias;
#pragma unroll
    for (int kt = 0; kt < KT; ++kt) {
      BF16x16 bfrag;
      const uint4* p =
          (const uint4*)(wf + (((size_t)(wave * 4 + nt) * KT + kt) * 32 + lane) * 8);
      bfrag.q[0] = p[0];
      bfrag.q[1] = p[1];
      acc = __builtin_amdgcn_wmma_f32_16x16x32_bf16(
          false, afrag[kt].v, false, bfrag.v, (short)0, acc, false, false);
    }
    *(v8f*)(pre_sw + (size_t)tile * 8192 + ((wave * 4 + nt) * 32 + lane) * 8) = acc;
  }
}

// ---------------------------------------------------------------------------
// Recurrence. grid.x = 16 batch tiles, block = 256 (8 waves), 512 steps.
// ---------------------------------------------------------------------------
#define GPAD 516
#define HPAD 132

__global__ __launch_bounds__(256) void lstm_recurrent_kernel(
    const float* __restrict__ pre_sw,      // swizzled, 32KB per (t,btile)
    const unsigned int* __restrict__ wfhh, // W_hh bf16 B-fragments
    unsigned int* __restrict__ af_out,     // next layer A-frags (or null)
    float* __restrict__ fout) {            // final output f32 (or null)
  __shared__ __align__(16) float gates[16][GPAD];
  __shared__ __align__(16) unsigned short hbuf[16][HPAD];
#if HAVE_TDM
  __shared__ __align__(32) float prebuf[2][8192];  // 2 x 32KB TDM double buffer
#endif

  const int tid = threadIdx.x;
  const int lane = tid & 31, wave = tid >> 5;
  const int r = lane & 15, half = lane >> 4;
  const int btile = blockIdx.x;

  // W_hh B-fragments resident in VGPRs for all 512 steps.
  BF16x16 wfrag[4][4];
#pragma unroll
  for (int nt = 0; nt < 4; ++nt)
#pragma unroll
    for (int kt = 0; kt < 4; ++kt) {
      const uint4* p =
          (const uint4*)(wfhh + (((size_t)(wave * 4 + nt) * 4 + kt) * 32 + lane) * 8);
      wfrag[nt][kt].q[0] = p[0];
      wfrag[nt][kt].q[1] = p[1];
    }

#if HAVE_TDM
  // Issue one Tensor-Data-Mover 1-D copy (32KB, f32) of this block's chunk.
  auto tdm_fetch = [&](int t, int buf) {
    const unsigned long long ga =
        (unsigned long long)(size_t)(pre_sw + ((size_t)t * 16 + btile) * 8192);
    // Flat LDS addresses carry the LDS byte offset in their low 32 bits.
    const unsigned lds = (unsigned)(size_t)(const void*)&prebuf[buf][0];
    u32x4 g0;
    g0[0] = 1u;  // count=1, user descriptor
    g0[1] = lds; // lds_addr [63:32]
    g0[2] = (unsigned)ga;                                  // global_addr lo
    g0[3] = (unsigned)((ga >> 32) & 0x01FFFFFFull) | (2u << 30);  // hi | type=2
    i32x8 g1;
    g1[0] = (int)(2u << 16);            // data_size = 4B, no multicast
    g1[1] = (int)(8192u << 16);         // tensor_dim0[15:0]
    g1[2] = (int)((8192u >> 16) | (1u << 16));  // tensor_dim0 hi, tensor_dim1=1
    g1[3] = (int)(8192u << 16);         // tile_dim0 = 8192 elements
    g1[4] = 0;
    g1[5] = (int)8192u;                 // tensor_dim0_stride
    g1[6] = 0;
    g1[7] = 0;
    i32x4 g2;
    g2[0] = 1; g2[1] = 0; g2[2] = 0; g2[3] = 0;  // tensor_dim2=1
    i32x4 g3;
    g3[0] = 0; g3[1] = 0; g3[2] = 0; g3[3] = 0;
    i32x8 g4;  // clang-23 toolchain: extra i32x8 group before cpol
    g4[0] = 0; g4[1] = 0; g4[2] = 0; g4[3] = 0;
    g4[4] = 0; g4[5] = 0; g4[6] = 0; g4[7] = 0;
    __builtin_amdgcn_tensor_load_to_lds(g0, g1, g2, g3, g4, 0);
  };
  if (wave == 0) tdm_fetch(0, 0);
#endif

  // Elementwise ownership: thread -> (row b, cols j0..j0+7); cell in regs.
  const int b = tid >> 4;
  const int j0 = (tid & 15) * 8;
  float c[8];
#pragma unroll
  for (int e = 0; e < 8; ++e) {
    c[e] = 0.0f;
    hbuf[b][j0 + e] = 0;
  }
#if HAVE_TDM
  if (wave == 0) __builtin_amdgcn_s_wait_tensorcnt(0);
#endif
  __syncthreads();

  for (int t = 0; t < TT; ++t) {
    const int p = t & 1;
#if HAVE_TDM
    if (wave == 0 && t + 1 < TT) tdm_fetch(t + 1, p ^ 1);  // prefetch next step
#endif
    // A fragments of h_{t-1} from LDS bf16.
    BF16x16 ah[4];
    const unsigned short* hrow = &hbuf[r][0];
#pragma unroll
    for (int kt = 0; kt < 4; ++kt)
#pragma unroll
      for (int v = 0; v < 8; ++v)
        ah[kt].u32[v] = *(const unsigned int*)(hrow + kt * 32 + a_koff(v, half));

#pragma unroll
    for (int nt = 0; nt < 4; ++nt) {
      const int g = wave * 64 + nt * 16 + r;
      v8f acc;
#if HAVE_TDM
      acc = *(const v8f*)&prebuf[p][((wave * 4 + nt) * 32 + lane) * 8];
#else
      acc = *(const v8f*)(pre_sw + ((size_t)t * 16 + btile) * 8192 +
                          ((wave * 4 + nt) * 32 + lane) * 8);
#endif
#pragma unroll
      for (int kt = 0; kt < 4; ++kt)
        acc = __builtin_amdgcn_wmma_f32_16x16x32_bf16(
            false, ah[kt].v, false, wfrag[nt][kt].v, (short)0, acc, false, false);
#pragma unroll
      for (int v = 0; v < 8; ++v) gates[8 * half + v][g] = acc[v];
    }
    __syncthreads();

    // Gates -> cell/hidden (PyTorch order i,f,g,o), native exp2/rcp.
    float hv[8];
#pragma unroll
    for (int e = 0; e < 8; ++e) {
      const int j = j0 + e;
      const float ig = sigmoidf_(gates[b][j]);
      const float fg = sigmoidf_(gates[b][j + HH]);
      const float gg = tanhf_(gates[b][j + 2 * HH]);
      const float og = sigmoidf_(gates[b][j + 3 * HH]);
      const float cn = fg * c[e] + ig * gg;
      c[e] = cn;
      hv[e] = og * tanhf_(cn);
      unsigned int u = __float_as_uint(hv[e]);  // h -> bf16 for next step
      u += 0x7FFFu + ((u >> 16) & 1u);
      hbuf[b][j] = (unsigned short)(u >> 16);
    }
    if (fout) {  // final layer: f32 row-major output
      float* hop = fout + ((size_t)t * BB + btile * 16 + b) * HH + j0;
#pragma unroll
      for (int e = 0; e < 8; ++e) hop[e] = hv[e];
    }
    if (af_out) {  // next layer input as bf16 A-fragments (one uint4 store)
      const int kt = j0 >> 5;
      const int hh = (j0 >> 3) & 1;
      const int v0 = ((j0 >> 4) & 1) << 2;
      const int ln = hh * 16 + b;
      uint4* dst = (uint4*)(af_out + (((size_t)(t * 16 + btile) * 4 + kt) * 32 + ln) * 8 + v0);
      *dst = make_uint4(pack2bf(hv[0], hv[1]), pack2bf(hv[2], hv[3]),
                        pack2bf(hv[4], hv[5]), pack2bf(hv[6], hv[7]));
    }
#if HAVE_TDM
    if (wave == 0) __builtin_amdgcn_s_wait_tensorcnt(0);
#endif
    __syncthreads();
  }
}

// ---------------------------------------------------------------------------
extern "C" void kernel_launch(void* const* d_in, const int* in_sizes, int n_in,
                              void* d_out, int out_size, void* d_ws,
                              size_t ws_size, hipStream_t stream) {
  (void)in_sizes; (void)n_in; (void)out_size; (void)ws_size;
  const float* x     = (const float*)d_in[0];  // [T,B,64]
  const float* W_ih0 = (const float*)d_in[1];  // [512,64]
  const float* W_ihr = (const float*)d_in[2];  // [4,512,128]
  const float* W_hh  = (const float*)d_in[3];  // [5,512,128]
  const float* b_ih  = (const float*)d_in[4];  // [5,512]
  const float* b_hh  = (const float*)d_in[5];  // [5,512]
  float* out = (float*)d_out;                  // [T,B,128]

  char* ws = (char*)d_ws;
  float* pre = (float*)ws;  // 256 MB
  unsigned int* afA = (unsigned int*)(ws + (size_t)TILES * 8192 * 4);
  unsigned int* afB = afA + (size_t)TILES * 4 * 32 * 8;       // +32 MB
  unsigned int* wfp = afB + (size_t)TILES * 4 * 32 * 8;       // +32 MB
  unsigned int* wf_ih0 = wfp;                                  // 64 KB
  unsigned int* wf_ih[5];
  wf_ih[0] = wf_ih0;
  for (int l = 1; l < 5; ++l)
    wf_ih[l] = wfp + 16384 + (size_t)(l - 1) * 32768;          // 128 KB each
  unsigned int* wf_hh[5];
  for (int l = 0; l < 5; ++l)
    wf_hh[l] = wfp + 16384 + 4 * 32768 + (size_t)l * 32768;

  // ---- one-time preps --------------------------------------------------
  x_to_afrag_kernel<<<TILES, 64, 0, stream>>>(x, afA);
  prep_wfrag_kernel<<<2, 256, 0, stream>>>(W_ih0, 64, wf_ih[0]);
  for (int l = 1; l < 5; ++l)
    prep_wfrag_kernel<<<4, 256, 0, stream>>>(W_ihr + (size_t)(l - 1) * GH * HH,
                                             128, wf_ih[l]);
  for (int l = 0; l < 5; ++l)
    prep_wfrag_kernel<<<4, 256, 0, stream>>>(W_hh + (size_t)l * GH * HH, 128,
                                             wf_hh[l]);

  // ---- layers ----------------------------------------------------------
  unsigned int* af_in = afA;
  unsigned int* af_nx = afB;
  for (int l = 0; l < 5; ++l) {
    if (l == 0)
      pre_gemm_kernel<64><<<TILES, 256, 0, stream>>>(af_in, wf_ih[0], b_ih,
                                                     b_hh, pre);
    else
      pre_gemm_kernel<128><<<TILES, 256, 0, stream>>>(
          af_in, wf_ih[l], b_ih + (size_t)l * GH, b_hh + (size_t)l * GH, pre);
    const bool last = (l == 4);
    lstm_recurrent_kernel<<<BB / 16, 256, 0, stream>>>(
        pre, wf_hh[l], last ? nullptr : af_nx, last ? out : nullptr);
    unsigned int* tmp = af_in; af_in = af_nx; af_nx = tmp;
  }
}
